// OptimizedMambaBlock_8881992368586
// MI455X (gfx1250) — matrix operand order repaired and checked by reading, not compiled, not run on hardware
//
#include <hip/hip_runtime.h>

// ---------------- CDNA5 WMMA types ----------------
typedef __attribute__((ext_vector_type(16))) __bf16 v16bf;
typedef __attribute__((ext_vector_type(8)))  __bf16 v8bf;
typedef __attribute__((ext_vector_type(8)))  float  v8f;
typedef int i32x4 __attribute__((__vector_size__(16)));

// ---------------- problem dims ----------------
#define D_MODEL 1024
#define D_STATE 16
#define D_CONV  4
#define D_INNER 2048
#define BATCH   2
#define LENGTH  1024
#define MROWS   (BATCH * LENGTH)        // 2048 token rows
#define NXZ     (2 * D_INNER)           // 4096
#define NDBU    (2 * D_STATE + D_INNER) // 2080

// ---------------- async-to-LDS capability probes ----------------
#if defined(__has_builtin)
#  if __has_builtin(__builtin_amdgcn_global_load_async_to_lds_b128)
#    define HAVE_ASYNC_LDS 1
#  else
#    define HAVE_ASYNC_LDS 0
#  endif
#  if __has_builtin(__builtin_amdgcn_s_wait_asynccnt)
#    define HAVE_WAIT_ASYNC 1
#  else
#    define HAVE_WAIT_ASYNC 0
#  endif
#else
#  define HAVE_ASYNC_LDS 0
#  define HAVE_WAIT_ASYNC 0
#endif

#if HAVE_ASYNC_LDS
#  if HAVE_WAIT_ASYNC
#    define WAIT_ASYNC(n) __builtin_amdgcn_s_wait_asynccnt(n)
#  else
#    define WAIT_ASYNC(n) asm volatile("s_wait_asynccnt " #n ::: "memory")
#  endif
#else
#  define WAIT_ASYNC(n) do {} while (0)
#endif

// copy 16 bytes global -> LDS (async if the CDNA5 path is available)
__device__ __forceinline__ void async_copy16(const __bf16* g, __bf16* l) {
#if HAVE_ASYNC_LDS
  __builtin_amdgcn_global_load_async_to_lds_b128(
      (__attribute__((address_space(1))) i32x4*)g,
      (__attribute__((address_space(3))) i32x4*)l, 0, 0);
#else
  *reinterpret_cast<v8bf*>(l) = *reinterpret_cast<const v8bf*>(g);
#endif
}

// ============================================================
// Prologue: LDS-tiled transpose-cast fp32 (K,N) -> bf16 (N,K).
// ============================================================
__global__ void mamba_wcast_t(const float* __restrict__ W, __bf16* __restrict__ Wt,
                              int K, int N) {
  __shared__ float tile[32][33];
  const int k0 = blockIdx.x * 32;
  const int n0 = blockIdx.y * 32;
#pragma unroll
  for (int j = 0; j < 32; j += 8)
    tile[threadIdx.y + j][threadIdx.x] =
        W[(long)(k0 + threadIdx.y + j) * N + n0 + threadIdx.x];
  __syncthreads();
#pragma unroll
  for (int j = 0; j < 32; j += 8)
    Wt[(long)(n0 + threadIdx.y + j) * K + k0 + threadIdx.x] =
        (__bf16)tile[threadIdx.x][threadIdx.y + j];
}

__global__ void mamba_fcast(const float* __restrict__ in, __bf16* __restrict__ out, long n) {
  long i = (long)blockIdx.x * blockDim.x + threadIdx.x;
  if (i < n) out[i] = (__bf16)in[i];
}

// ============================================================
// LDS-staged bf16 WMMA GEMM (N % 256 == 0, M % 64 == 0, K % 32 == 0).
// Block = 8 waves (2x4); block tile 64x256, wave tile 32x64.
// Per K=32 stage: async-copy A(64x32)+B(256x32) into LDS
// (double-buffered, ASYNCcnt), then 8 WMMAs per wave from
// conflict-free ds_load_b128 reads (80B padded rows).
// ============================================================
#define BM 64
#define BN 256
#define BK 32
#define ROWB 80                       // 64B data + 16B pad; banks 20*r%64 distinct
#define ABYTES (BM * ROWB)            // 5120
#define BBYTES (BN * ROWB)            // 20480
#define STAGEB (ABYTES + BBYTES)      // 25600

__global__ void mamba_gemm_staged(const __bf16* __restrict__ A,
                                  const __bf16* __restrict__ Bt,
                                  const float* __restrict__ bias,
                                  float* __restrict__ C,
                                  int M, int N, int K) {
  __shared__ __align__(16) char smem[2 * STAGEB];
  const int t    = threadIdx.x;
  const int lane = t & 31, wave = t >> 5;
  const int wm   = wave >> 2, wn = wave & 3;
  const int half = lane >> 4, l16 = lane & 15;
  const int m0   = blockIdx.y * BM;
  const int n0   = blockIdx.x * BN;

  auto issue = [&](int s, int buf) {
    char* sb = smem + buf * STAGEB;
    const long kO = (long)s * BK;
    {   // A: 64 rows x 4 segs of 16B -> 1 per thread
      int row = t >> 2, seg = t & 3;
      async_copy16(A + (long)(m0 + row) * K + kO + seg * 8,
                   (__bf16*)(sb + row * ROWB + seg * 16));
    }
#pragma unroll
    for (int r = 0; r < 4; ++r) {  // B: 256 rows x 4 segs -> 4 per thread
      int idx = t + r * 256;
      int row = idx >> 2, seg = idx & 3;
      async_copy16(Bt + (long)(n0 + row) * K + kO + seg * 8,
                   (__bf16*)(sb + ABYTES + row * ROWB + seg * 16));
    }
  };

  v8f acc[2][4];
#pragma unroll
  for (int i = 0; i < 2; ++i)
#pragma unroll
    for (int j = 0; j < 4; ++j) acc[i][j] = (v8f){};

  const int stages = K / BK;
  issue(0, 0);
  for (int s = 0; s < stages; ++s) {
    if (s + 1 < stages) {
      issue(s + 1, (s + 1) & 1);
      WAIT_ASYNC(5);                 // stage s's 5 per-thread copies done
    } else {
      WAIT_ASYNC(0);
    }
    __syncthreads();                 // all waves' stage-s data visible

    const char* sb    = smem + (s & 1) * STAGEB;
    const char* bbase = sb + ABYTES;

    union { v16bf v; v8bf h[2]; } a[2], b[4];
#pragma unroll
    for (int i = 0; i < 2; ++i) {
      const char* ar = sb + (wm * 32 + i * 16 + l16) * ROWB;
      a[i].h[0] = *reinterpret_cast<const v8bf*>(ar + half * 16);
      a[i].h[1] = *reinterpret_cast<const v8bf*>(ar + 32 + half * 16);
    }
#pragma unroll
    for (int j = 0; j < 4; ++j) {
      const char* br = bbase + (wn * 64 + j * 16 + l16) * ROWB;
      b[j].h[0] = *reinterpret_cast<const v8bf*>(br + half * 16);
      b[j].h[1] = *reinterpret_cast<const v8bf*>(br + 32 + half * 16);
    }
#pragma unroll
    for (int i = 0; i < 2; ++i)
#pragma unroll
      for (int j = 0; j < 4; ++j)
        acc[i][j] = __builtin_amdgcn_wmma_f32_16x16x32_bf16(
            false, a[i].v, false, b[j].v, (short)0, acc[i][j], false, false);

    __syncthreads();                 // protect buffer reuse by next issue
  }

#pragma unroll
  for (int i = 0; i < 2; ++i) {
#pragma unroll
    for (int j = 0; j < 4; ++j) {
      const int col = n0 + wn * 64 + j * 16 + l16;
      const float bcol = bias[col];
      float* cptr = C + (long)(m0 + wm * 32 + i * 16 + half * 8) * N + col;
#pragma unroll
      for (int r = 0; r < 8; ++r)
        cptr[(long)r * N] = acc[i][j][r] + bcol;
    }
  }
}

// ============================================================
// Direct-load register-blocked WMMA GEMM (N % 32 == 0; used for
// the N=2080 x_proj GEMM). Wave tile (MT*16)x(NT*16).
// ============================================================
template <int MT, int NT>
__global__ void mamba_gemm_bf16(const __bf16* __restrict__ A,
                                const __bf16* __restrict__ Bt,
                                const float* __restrict__ bias,
                                float* __restrict__ C,
                                int M, int N, int K) {
  const int lane   = threadIdx.x & 31;
  const int wave   = threadIdx.x >> 5;
  const int tilesN = N / (NT * 16);
  const int tilesM = M / (MT * 16);
  const int tile   = blockIdx.x * 8 + wave;
  if (tile >= tilesM * tilesN) return;          // whole-wave exit; EXEC all-1s
  const int mt   = tile / tilesN;
  const int nt   = tile % tilesN;
  const int half = lane >> 4;
  const int l16  = lane & 15;

  const __bf16* arow[MT];
  const __bf16* brow[NT];
#pragma unroll
  for (int i = 0; i < MT; ++i)
    arow[i] = A + (long)(mt * MT * 16 + i * 16 + l16) * K;
#pragma unroll
  for (int j = 0; j < NT; ++j)
    brow[j] = Bt + (long)(nt * NT * 16 + j * 16 + l16) * K;

  v8f acc[MT][NT];
#pragma unroll
  for (int i = 0; i < MT; ++i)
#pragma unroll
    for (int j = 0; j < NT; ++j) acc[i][j] = (v8f){};

  for (int kk = 0; kk < K; kk += 32) {
    union { v16bf v; v8bf h[2]; } a[MT], b[NT];
#pragma unroll
    for (int i = 0; i < MT; ++i) {
      a[i].h[0] = *reinterpret_cast<const v8bf*>(arow[i] + kk + half * 8);
      a[i].h[1] = *reinterpret_cast<const v8bf*>(arow[i] + kk + 16 + half * 8);
    }
#pragma unroll
    for (int j = 0; j < NT; ++j) {
      b[j].h[0] = *reinterpret_cast<const v8bf*>(brow[j] + kk + half * 8);
      b[j].h[1] = *reinterpret_cast<const v8bf*>(brow[j] + kk + 16 + half * 8);
    }
#pragma unroll
    for (int i = 0; i < MT; ++i)
#pragma unroll
      for (int j = 0; j < NT; ++j)
        acc[i][j] = __builtin_amdgcn_wmma_f32_16x16x32_bf16(
            false, a[i].v, false, b[j].v, (short)0, acc[i][j], false, false);
  }

#pragma unroll
  for (int i = 0; i < MT; ++i) {
#pragma unroll
    for (int j = 0; j < NT; ++j) {
      const int col = nt * NT * 16 + j * 16 + l16;
      const float bcol = bias[col];
      float* cptr = C + (long)(mt * MT * 16 + i * 16 + half * 8) * N + col;
#pragma unroll
      for (int r = 0; r < 8; ++r)
        cptr[(long)r * N] = acc[i][j][r] + bcol;
    }
  }
}

// ============================================================
// Depthwise causal conv1d (width 4) + bias + SiLU -> bf16.
// ============================================================
__global__ void mamba_conv_silu(const float* __restrict__ xz,
                                const float* __restrict__ conv_w,
                                const float* __restrict__ conv_b,
                                __bf16* __restrict__ xc_act) {
  long i = (long)blockIdx.x * blockDim.x + threadIdx.x;
  if (i >= (long)MROWS * D_INNER) return;
  int d   = (int)(i % D_INNER);
  int row = (int)(i / D_INNER);
  int l   = row % LENGTH;
  int b   = row / LENGTH;

  float acc = conv_b[d];
#pragma unroll
  for (int j = 0; j < D_CONV; ++j) {
    int ll = l - (D_CONV - 1) + j;
    if (ll >= 0)
      acc += conv_w[d * D_CONV + j] * xz[((long)(b * LENGTH + ll)) * NXZ + d];
  }
  float s = acc / (1.0f + __expf(-acc));
  xc_act[(long)row * D_INNER + d] = (__bf16)s;
}

// ============================================================
// Selective scan, one lane per (batch, channel); fused dt_proj
// (hoisted weights), softplus, e^n recurrence, +u*D, silu(z) gate.
// ============================================================
__global__ void mamba_scan(const float* __restrict__ dBu,
                           const float* __restrict__ xz,
                           const float* __restrict__ dtw,
                           const float* __restrict__ dtb,
                           const float* __restrict__ Dp,
                           __bf16* __restrict__ ygate) {
  int g = blockIdx.x * blockDim.x + threadIdx.x;
  if (g >= BATCH * D_INNER) return;
  int b = g / D_INNER;
  int d = g % D_INNER;

  float h[D_STATE], w[D_STATE];
#pragma unroll
  for (int n = 0; n < D_STATE; ++n) {
    h[n] = 0.0f;
    w[n] = dtw[n * D_INNER + d];
  }

  const float dtbias = dtb[d];
  const float Dd     = Dp[d];

  for (int t = 0; t < LENGTH; ++t) {
    const long   row = (long)b * LENGTH + t;
    const float* dr  = dBu + row * NDBU;

    float s = dtbias;
#pragma unroll
    for (int k = 0; k < D_STATE; ++k) s += dr[k] * w[k];
    float delta = (s > 20.0f) ? s : log1pf(__expf(s));

    const float u   = dr[2 * D_STATE + d];
    const float e   = __expf(-delta);
    const float dbu = delta * u;

    float y = 0.0f, p = 1.0f;
#pragma unroll
    for (int n = 0; n < D_STATE; ++n) {
      p *= e;
      h[n] = p * h[n] + dbu * dr[D_STATE + n];
      y += h[n];
    }

    y += u * Dd;
    const float zv = xz[row * NXZ + D_INNER + d];
    const float gz = zv / (1.0f + __expf(-zv));
    ygate[row * D_INNER + d] = (__bf16)(y * gz);
  }
}

// ============================================================
// Host-side launch
// ============================================================
extern "C" void kernel_launch(void* const* d_in, const int* in_sizes, int n_in,
                              void* d_out, int out_size, void* d_ws, size_t ws_size,
                              hipStream_t stream) {
  (void)in_sizes; (void)n_in; (void)out_size; (void)ws_size;

  const float* x          = (const float*)d_in[0];
  const float* in_proj_w  = (const float*)d_in[1];
  const float* in_proj_b  = (const float*)d_in[2];
  const float* conv_w     = (const float*)d_in[3];
  const float* conv_b     = (const float*)d_in[4];
  const float* x_proj_w   = (const float*)d_in[5];
  const float* x_proj_b   = (const float*)d_in[6];
  const float* dt_proj_w  = (const float*)d_in[7];
  const float* dt_proj_b  = (const float*)d_in[8];
  const float* Dp         = (const float*)d_in[9];
  const float* out_proj_w = (const float*)d_in[10];
  const float* out_proj_b = (const float*)d_in[11];
  float* out = (float*)d_out;

  size_t off = 0;
  auto carve = [&](size_t bytes) {
    void* p = (char*)d_ws + off;
    off = (off + bytes + 255) & ~(size_t)255;
    return p;
  };
  __bf16* w_in_t  = (__bf16*)carve((size_t)NXZ * D_MODEL * 2);
  __bf16* w_x_t   = (__bf16*)carve((size_t)NDBU * D_INNER * 2);
  __bf16* w_out_t = (__bf16*)carve((size_t)D_MODEL * D_INNER * 2);
  __bf16* x_bf    = (__bf16*)carve((size_t)MROWS * D_MODEL * 2);
  float*  xz      = (float*) carve((size_t)MROWS * NXZ * 4);
  __bf16* xc_act  = (__bf16*)carve((size_t)MROWS * D_INNER * 2);
  float*  dBu     = (float*) carve((size_t)MROWS * NDBU * 4);
  __bf16* ygate   = (__bf16*)carve((size_t)MROWS * D_INNER * 2);

  const int TB = 256;
  auto blocks = [](long n, int tb) { return (int)((n + tb - 1) / tb); };

  // 1) weight transpose-casts (LDS-tiled) + input cast
  {
    dim3 tb(32, 8);
    mamba_wcast_t<<<dim3(D_MODEL / 32, NXZ / 32), tb, 0, stream>>>(in_proj_w, w_in_t, D_MODEL, NXZ);
    mamba_wcast_t<<<dim3(D_INNER / 32, NDBU / 32), tb, 0, stream>>>(x_proj_w, w_x_t, D_INNER, NDBU);
    mamba_wcast_t<<<dim3(D_INNER / 32, D_MODEL / 32), tb, 0, stream>>>(out_proj_w, w_out_t, D_INNER, D_MODEL);
    mamba_fcast<<<blocks((long)MROWS * D_MODEL, TB), TB, 0, stream>>>(x, x_bf, (long)MROWS * D_MODEL);
  }

  // 2) in_proj GEMM (M=2048,N=4096,K=1024): LDS-staged async pipeline
  mamba_gemm_staged<<<dim3(NXZ / BN, MROWS / BM), TB, 0, stream>>>(
      x_bf, w_in_t, in_proj_b, xz, MROWS, NXZ, D_MODEL);

  // 3) depthwise causal conv + silu -> bf16
  mamba_conv_silu<<<blocks((long)MROWS * D_INNER, TB), TB, 0, stream>>>(xz, conv_w, conv_b, xc_act);

  // 4) x_proj GEMM (M=2048,N=2080,K=2048): direct-load 32x32/wave
  {
    int tiles = (MROWS / 32) * (NDBU / 32);
    mamba_gemm_bf16<2, 2><<<(tiles + 7) / 8, TB, 0, stream>>>(xc_act, w_x_t, x_proj_b, dBu,
                                                              MROWS, NDBU, D_INNER);
  }

  // 5) selective scan (fused dt_proj / softplus / recurrence / gate)
  mamba_scan<<<(BATCH * D_INNER + 127) / 128, 128, 0, stream>>>(dBu, xz, dt_proj_w,
                                                                dt_proj_b, Dp, ygate);

  // 6) out_proj GEMM (M=2048,N=1024,K=2048): LDS-staged async pipeline
  mamba_gemm_staged<<<dim3(D_MODEL / BN, MROWS / BM), TB, 0, stream>>>(
      ygate, w_out_t, out_proj_b, out, MROWS, D_MODEL, D_INNER);
}
